// NoiseConv_35433480192229
// MI455X (gfx1250) — compile-verified, matrix operand order checked by reading, not run on hardware
//
#include <hip/hip_runtime.h>

typedef __attribute__((ext_vector_type(16))) _Float16 v16h;
typedef __attribute__((ext_vector_type(8)))  _Float16 h8;
typedef __attribute__((ext_vector_type(8)))  float    v8f;
typedef __attribute__((ext_vector_type(4)))  float    f4;

#define NB      8
#define C_IN    256
#define C_OUT   256
#define HW      4096
#define NPIX    (NB * HW)        // 32768 pixels
#define TN      64               // N-tile per workgroup
#define KC      32               // K chunk per WMMA

union AFrag { v16h v; h8 h[2]; };

// ---------------------------------------------------------------------------
// Kernel 1: per-pixel channel min/max -> cmin, scale, 1/scale
// ---------------------------------------------------------------------------
__global__ __launch_bounds__(256) void stats_kernel(const float* __restrict__ x,
                                                    float* __restrict__ cmin,
                                                    float* __restrict__ scale,
                                                    float* __restrict__ rscale) {
  int pix = blockIdx.x * 256 + threadIdx.x;      // 0..32767
  int b   = pix >> 12;
  int hw  = pix & (HW - 1);
  const float* xp = x + (size_t)b * C_IN * HW + hw;
  float mn = xp[0];
  float mx = mn;
#pragma unroll 4
  for (int c = 1; c < C_IN; ++c) {
    float v = xp[(size_t)c * HW];
    mn = fminf(mn, v);
    mx = fmaxf(mx, v);
  }
  float s = (mx - mn) * (1.0f / 63.0f);
  cmin[pix]   = mn;
  scale[pix]  = s;
  rscale[pix] = (s > 0.0f) ? (1.0f / s) : 0.0f;
}

// ---------------------------------------------------------------------------
// Kernel 2: weight fp32 -> fp16 copy + per-row sum
// ---------------------------------------------------------------------------
__global__ __launch_bounds__(256) void wprep_kernel(const float* __restrict__ w,
                                                    _Float16* __restrict__ wh,
                                                    float* __restrict__ wsum) {
  __shared__ float red[256];
  int o = blockIdx.x;      // output channel
  int c = threadIdx.x;     // input channel
  float v = w[o * C_IN + c];
  wh[o * C_IN + c] = (_Float16)v;
  red[c] = v;
  __syncthreads();
#pragma unroll
  for (int s = 128; s > 0; s >>= 1) {
    if (c < s) red[c] += red[c + s];
    __syncthreads();
  }
  if (c == 0) wsum[o] = red[0];
}

// ---------------------------------------------------------------------------
// Kernel 3: fused quantize + 1x1-conv GEMM via v_wmma_f32_16x16x32_f16
//   Workgroup: full M=256 x N=64 tile. 8 waves; wave w owns M rows
//   [32w, 32w+32) x all 64 N cols -> 2x4 fragments of 16x16.
// ---------------------------------------------------------------------------
__global__ __launch_bounds__(256) void gemm_kernel(const float* __restrict__ x,
                                                   const float* __restrict__ bias,
                                                   const float* __restrict__ cmin,
                                                   const float* __restrict__ scale,
                                                   const float* __restrict__ rscale,
                                                   const float* __restrict__ wsum,
                                                   const _Float16* __restrict__ wh,
                                                   float* __restrict__ out) {
  __shared__ _Float16 As[C_OUT * KC];   // [M][32] f16 weight chunk (16 KB)
  __shared__ _Float16 Bs[TN * KC];      // [N][32] f16 quantized x chunk (4 KB)
  __shared__ float sCmin[TN], sScale[TN], sRs[TN];

  const int t    = threadIdx.x;   // 0..255
  const int lane = t & 31;
  const int wave = t >> 5;        // 0..7
  const int n0   = blockIdx.x * TN;       // global pixel base (64 | 4096)
  const int b    = n0 >> 12;
  const int hw0  = n0 & (HW - 1);

  // stage per-pixel stats for this N tile
  if (t < TN) {
    sCmin[t]  = cmin[n0 + t];
    sScale[t] = scale[n0 + t];
  } else if (t < 2 * TN) {
    sRs[t - TN] = rscale[n0 + t - TN];
  }

  const v8f vzero = {0.f, 0.f, 0.f, 0.f, 0.f, 0.f, 0.f, 0.f};
  v8f acc[2][4];
#pragma unroll
  for (int mi = 0; mi < 2; ++mi)
#pragma unroll
    for (int ni = 0; ni < 4; ++ni)
      acc[mi][ni] = vzero;

  // B-chunk loader mapping: thread t -> channel-local cl (0..31), pixel j0..j0+7
  const int cl = t >> 3;
  const int j0 = (t & 7) * 8;
  const float* xb = x + (size_t)b * C_IN * HW + hw0 + j0;

  const int lm = lane & 15;            // M/N within tile
  const int bk = (lane >> 4) * 8;      // K base: 0 or 8 (ISA 16-bit A/B layout)

  __syncthreads();   // stats staged

  for (int kc = 0; kc < C_IN / KC; ++kc) {
    const int kk = kc * KC;

    // ---- global loads into registers ----
    const h8* wrow = (const h8*)(wh + (size_t)t * C_IN + kk);  // 16B aligned
    h8 a0 = wrow[0], a1 = wrow[1], a2 = wrow[2], a3 = wrow[3];

    const float* xp = xb + (size_t)(kk + cl) * HW;
    f4 xv0 = *(const f4*)(xp + 0);
    f4 xv1 = *(const f4*)(xp + 4);

    // ---- quantize: x_int = rint((v - cmin)*rscale), exact in f16 ----
    _Float16 q[8];
#pragma unroll
    for (int i = 0; i < 4; ++i)
      q[i] = (_Float16)rintf((xv0[i] - sCmin[j0 + i]) * sRs[j0 + i]);
#pragma unroll
    for (int i = 0; i < 4; ++i)
      q[4 + i] = (_Float16)rintf((xv1[i] - sCmin[j0 + 4 + i]) * sRs[j0 + 4 + i]);

    __syncthreads();   // previous iteration's LDS reads complete

    // As[M=t][0..31]
    h8* arow = (h8*)&As[t * KC];
    arow[0] = a0; arow[1] = a1; arow[2] = a2; arow[3] = a3;
    // Bs[N][K] transpose-store
#pragma unroll
    for (int i = 0; i < 8; ++i)
      Bs[(j0 + i) * KC + cl] = q[i];

    __syncthreads();   // chunk staged

    // ---- fragments: elements 0-7 -> K=bk+0..7, elements 8-15 -> K=bk+16..23
    AFrag af[2], bf[4];
#pragma unroll
    for (int mi = 0; mi < 2; ++mi) {
      int M = wave * 32 + mi * 16 + lm;
      af[mi].h[0] = *(const h8*)&As[M * KC + bk];
      af[mi].h[1] = *(const h8*)&As[M * KC + 16 + bk];
    }
#pragma unroll
    for (int ni = 0; ni < 4; ++ni) {
      int N = ni * 16 + lm;
      bf[ni].h[0] = *(const h8*)&Bs[N * KC + bk];
      bf[ni].h[1] = *(const h8*)&Bs[N * KC + 16 + bk];
    }

#pragma unroll
    for (int mi = 0; mi < 2; ++mi)
#pragma unroll
      for (int ni = 0; ni < 4; ++ni)
        acc[mi][ni] = __builtin_amdgcn_wmma_f32_16x16x32_f16(
            false, af[mi].v, false, bf[ni].v,
            (short)0, acc[mi][ni], false, false);
  }

  // ---- epilogue: out = acc*scale + cmin*wsum + bias ----
  // D layout: lane -> N=lm, VGPR r -> M = r + 8*(lane>=16)
  const int mh = (lane >> 4) * 8;
#pragma unroll
  for (int mi = 0; mi < 2; ++mi) {
#pragma unroll
    for (int ni = 0; ni < 4; ++ni) {
      int N = ni * 16 + lm;
      float sc = sScale[N];
      float cm = sCmin[N];
      size_t col = (size_t)b * C_OUT * HW + hw0 + N;
#pragma unroll
      for (int r = 0; r < 8; ++r) {
        int M = wave * 32 + mi * 16 + mh + r;
        float v = acc[mi][ni][r] * sc + cm * wsum[M] + bias[M];
        out[col + (size_t)M * HW] = v;
      }
    }
  }
}

// ---------------------------------------------------------------------------
extern "C" void kernel_launch(void* const* d_in, const int* in_sizes, int n_in,
                              void* d_out, int out_size, void* d_ws, size_t ws_size,
                              hipStream_t stream) {
  (void)in_sizes; (void)n_in; (void)out_size; (void)ws_size;
  const float* x    = (const float*)d_in[0];   // [8,256,64,64]
  const float* w    = (const float*)d_in[1];   // [256,256]
  const float* bias = (const float*)d_in[2];   // [256]
  float* out = (float*)d_out;                  // [8,256,64,64]

  float* ws      = (float*)d_ws;
  float* cmin    = ws;                         // 32768 f32
  float* scale   = cmin + NPIX;                // 32768 f32
  float* rscale  = scale + NPIX;               // 32768 f32
  float* wsum    = rscale + NPIX;              // 256 f32
  _Float16* wh   = (_Float16*)(wsum + 256);    // 65536 f16 (16B-aligned offset)

  stats_kernel<<<NPIX / 256, 256, 0, stream>>>(x, cmin, scale, rscale);
  wprep_kernel<<<C_OUT, 256, 0, stream>>>(w, wh, wsum);
  gemm_kernel<<<NPIX / TN, 256, 0, stream>>>(x, bias, cmin, scale, rscale,
                                             wsum, wh, out);
}
